// Attention_52046413693513
// MI455X (gfx1250) — matrix operand order repaired and checked by reading, not compiled, not run on hardware
//
#include <hip/hip_runtime.h>

#define DEV __device__ __forceinline__

typedef __bf16 bf16;
typedef __bf16 v16bf __attribute__((ext_vector_type(16)));
typedef __bf16 v8bf  __attribute__((ext_vector_type(8)));
typedef float  v8f   __attribute__((ext_vector_type(8)));
typedef unsigned int u32x4 __attribute__((ext_vector_type(4)));
typedef int          i32x4 __attribute__((ext_vector_type(4)));
typedef int          i32x8 __attribute__((ext_vector_type(8)));

union Frag16 { v16bf v; v8bf h[2]; };
union F8     { v8f   v; float f[8]; };

static constexpr int   BATCH  = 2;
static constexpr int   NSEQ   = 2048;
static constexpr int   DIN    = 1024;   // == INNER
static constexpr int   NHEAD  = 16;
static constexpr int   DH     = 64;
static constexpr float SCALE  = 0.125f; // 64^-0.5
static constexpr float LN_EPS = 1e-5f;

#if defined(__gfx1250__) && __has_builtin(__builtin_amdgcn_tensor_load_to_lds) \
    && __has_builtin(__builtin_amdgcn_s_wait_tensorcnt)
#define ATTN_USE_TDM 1
#else
#define ATTN_USE_TDM 0
#endif

// ---------------- WMMA helpers ----------------

DEV v8f wmma_bf16(v16bf a, v16bf b, v8f c) {
  // D = A(16x32) * B(32x16) + C, f32 accumulate
  return __builtin_amdgcn_wmma_f32_16x16x32_bf16(
      /*neg_a=*/false, a, /*neg_b=*/false, b,
      /*c_mod=*/(short)0, c, /*reuse_a=*/false, /*reuse_b=*/false);
}

// A-matrix fragment (16x32 bf16). p = row_base + k + (lane/16)*8.
// elements [0..7] = p[0..7], [8..15] = p[16..23].
DEV v16bf ld_fragA(const bf16* p) {
  Frag16 f;
  f.h[0] = *(const v8bf*)(p);
  f.h[1] = *(const v8bf*)(p + 16);
  return f.v;
}

// B-matrix fragment (32x16 bf16), K-contiguous column source.
// p = col_base + k + (lane/16)*8 ; elements [0..7] = p[0..7], [8..15] = p[16..23].
DEV v16bf ld_fragB(const bf16* p) {
  Frag16 f;
  f.h[0] = *(const v8bf*)(p);
  f.h[1] = *(const v8bf*)(p + 16);
  return f.v;
}

#if ATTN_USE_TDM
// Issue a TDM load of a 64x64 bf16 tile (row stride in elements) into LDS.
// D# per CDNA5 ISA ch.8: group0 = count/lds_addr/global_addr/type,
// group1 = data_size + dims/strides, remaining groups unused for 2-D tiles.
// This toolchain exposes the 6-arg builtin:
//   (u32x4 g0, i32x8 g1, i32x4 g2, i32x4 g3, i32x8 g4, i32 cpol)
DEV void tdm_load_tile_64x64(const bf16* gsrc, unsigned lds_off,
                             unsigned row_stride_elems) {
  const unsigned long long ga = (unsigned long long)(const void*)gsrc;
  u32x4 g0;
  g0[0] = 1u;                                   // count=1 (valid user D#)
  g0[1] = lds_off;                              // lds_addr (bytes)
  g0[2] = (unsigned)ga;                         // global_addr[31:0]
  g0[3] = (unsigned)(ga >> 32) | 0x80000000u;   // addr[56:32] | type=2<<30
  i32x8 g1;
  g1[0] = 0x00010000;          // wg_mask=0, data_size=1 (2 bytes)
  g1[1] = (int)(64u << 16);    // tensor_dim0[15:0]=64 in bits 63:48
  g1[2] = (int)(64u << 16);    // tensor_dim0 hi=0 ; tensor_dim1[15:0]=64
  g1[3] = (int)(64u << 16);    // tensor_dim1 hi=0 ; tile_dim0=64
  g1[4] = 64;                  // tile_dim1=64 ; tile_dim2=0 (2-D)
  g1[5] = (int)row_stride_elems; // tensor_dim0_stride[31:0]
  g1[6] = 0;                   // stride hi / tensor_dim1_stride lo
  g1[7] = 0;
  i32x4 z4 = {};
  i32x8 z8 = {};
  __builtin_amdgcn_tensor_load_to_lds(g0, g1, z4, z4, z8, 0);
}
#endif

// ---------------- Kernel 1: LayerNorm + bf16 cast ----------------
// grid (4096, 3), block 256.  out[t][row][k] bf16.
__global__ __launch_bounds__(256)
void ln_cast_kernel(const float* __restrict__ q, const float* __restrict__ k,
                    const float* __restrict__ v, const float* __restrict__ g,
                    const float* __restrict__ bias, bf16* __restrict__ out) {
  const int row = blockIdx.x;
  const int t   = blockIdx.y;
  const float* src = (t == 0) ? q : (t == 1) ? k : v;
  src += (size_t)row * DIN;
  bf16* dst = out + ((size_t)t * BATCH * NSEQ + row) * DIN;

  float x[4];
  float s = 0.f, s2 = 0.f;
#pragma unroll
  for (int i = 0; i < 4; ++i) {
    x[i] = src[threadIdx.x + i * 256];
    s += x[i];
    s2 += x[i] * x[i];
  }
  __shared__ float red0[256];
  __shared__ float red1[256];
  red0[threadIdx.x] = s;
  red1[threadIdx.x] = s2;
  __syncthreads();
  for (int off = 128; off > 0; off >>= 1) {
    if (threadIdx.x < off) {
      red0[threadIdx.x] += red0[threadIdx.x + off];
      red1[threadIdx.x] += red1[threadIdx.x + off];
    }
    __syncthreads();
  }
  const float mu  = red0[0] * (1.f / DIN);
  const float var = red1[0] * (1.f / DIN) - mu * mu;
  const float rs  = rsqrtf(var + LN_EPS);
#pragma unroll
  for (int i = 0; i < 4; ++i) {
    const int c = threadIdx.x + i * 256;
    dst[c] = (bf16)((x[i] - mu) * rs * g[c] + bias[c]);
  }
}

// ---------------- Kernel 2: weight cast + transpose ----------------
// Wt[n][k] = w[k][n], fp32 -> bf16.  grid (4096, 4), block 256.
__global__ __launch_bounds__(256)
void wcast_kernel(const float* __restrict__ wq, const float* __restrict__ wk,
                  const float* __restrict__ wv, const float* __restrict__ wo,
                  bf16* __restrict__ out) {
  const int w = blockIdx.y;
  const float* src = (w == 0) ? wq : (w == 1) ? wk : (w == 2) ? wv : wo;
  bf16* dst = out + (size_t)w * DIN * DIN;
  const int idx = blockIdx.x * 256 + threadIdx.x;
  const int n = idx >> 10;
  const int k = idx & 1023;
  dst[idx] = (bf16)src[(size_t)k * DIN + n];
}

// ---------------- Kernel 3: bf16 WMMA GEMM ----------------
// C[m][n] = sum_k A[m][k] * Bt[n][k] + bias[n]
// mode 0: bf16 out row-major; mode 1: bf16 out transposed-per-head (Vt);
// mode 2: fp32 out row-major.
// block 128 (4 waves, 2x2 of 64x64 wave tiles), grid (8, 32).
__global__ __launch_bounds__(128)
void gemm_kernel(const bf16* __restrict__ A, const bf16* __restrict__ Bt,
                 const float* __restrict__ bias, void* __restrict__ out,
                 int mode) {
  const int lane = threadIdx.x & 31;
  const int wid  = threadIdx.x >> 5;
  const int r    = lane & 15;
  const int hf   = lane >> 4;
  const int m0   = blockIdx.y * 128 + (wid >> 1) * 64;
  const int n0   = blockIdx.x * 128 + (wid & 1) * 64;

  v8f zero = {};
  F8 acc[4][4];
#pragma unroll
  for (int i = 0; i < 4; ++i)
#pragma unroll
    for (int j = 0; j < 4; ++j) acc[i][j].v = zero;

  const bf16* ap[4];
  const bf16* bp[4];
#pragma unroll
  for (int i = 0; i < 4; ++i) {
    ap[i] = A  + (size_t)(m0 + i * 16 + r) * DIN + hf * 8;
    bp[i] = Bt + (size_t)(n0 + i * 16 + r) * DIN + hf * 8;
  }

  for (int k = 0; k < DIN; k += 32) {
    v16bf af[4], bfr[4];
#pragma unroll
    for (int i = 0; i < 4; ++i) {
      af[i] = ld_fragA(ap[i] + k);
      __builtin_prefetch(ap[i] + k + 128, 0, 1);
    }
#pragma unroll
    for (int j = 0; j < 4; ++j) {
      bfr[j] = ld_fragB(bp[j] + k);
      __builtin_prefetch(bp[j] + k + 128, 0, 1);
    }
#pragma unroll
    for (int i = 0; i < 4; ++i)
#pragma unroll
      for (int j = 0; j < 4; ++j)
        acc[i][j].v = wmma_bf16(af[i], bfr[j], acc[i][j].v);
  }

#pragma unroll
  for (int j = 0; j < 4; ++j) {
    const int col = n0 + j * 16 + r;
    const float bn = bias[col];
#pragma unroll
    for (int i = 0; i < 4; ++i) {
#pragma unroll
      for (int v = 0; v < 8; ++v) {
        const int row = m0 + i * 16 + v + hf * 8;
        const float val = acc[i][j].f[v] + bn;
        if (mode == 0) {
          ((bf16*)out)[(size_t)row * DIN + col] = (bf16)val;
        } else if (mode == 1) {
          // Vt[((b*16 + h)*64 + d)][n_local] = V[row][col]
          const int b  = row >> 11;
          const int nl = row & (NSEQ - 1);
          const int h  = col >> 6;
          const int d  = col & 63;
          ((bf16*)out)[(((size_t)(b * NHEAD + h) * DH + d) << 11) + nl] =
              (bf16)val;
        } else {
          ((float*)out)[(size_t)row * DIN + col] = val;
        }
      }
    }
  }
}

// ---------------- Kernel 4: flash attention ----------------
// grid (NSEQ/128, NHEAD, BATCH), block 256 (8 waves, 16 query rows each).
// K/V tiles staged in LDS by the Tensor Data Mover (double buffered);
// output written directly in the "faithful reshape" scrambled layout.
__global__ __launch_bounds__(256)
void attn_kernel(const bf16* __restrict__ Qp, const bf16* __restrict__ Kp,
                 const bf16* __restrict__ Vt, bf16* __restrict__ Os) {
  const int lane = threadIdx.x & 31;
  const int w    = threadIdx.x >> 5;
  const int r    = lane & 15;
  const int hf   = lane >> 4;
  const int b    = blockIdx.z;
  const int h    = blockIdx.y;
  const int m0   = blockIdx.x * 128 + w * 16;  // query row within batch

  __shared__ __align__(16) bf16 kbuf[2][64 * 64];  // [n_local][d]
  __shared__ __align__(16) bf16 vbuf[2][64 * 64];  // [d][n_local]
  __shared__ __align__(16) bf16 pbuf[8][16][64];

  // Q fragments for 16 rows, K-dim 64 -> two 16x32 A-frags
  const bf16* qrow = Qp + ((size_t)(b * NSEQ + m0 + r)) * DIN + h * DH + hf * 8;
  v16bf aQ0 = ld_fragA(qrow);
  v16bf aQ1 = ld_fragA(qrow + 32);

  const bf16* kg = Kp + (size_t)b * NSEQ * DIN + h * DH;          // row = n
  const bf16* vg = Vt + ((size_t)(b * NHEAD + h) * DH) * NSEQ;    // row = d

  v8f zero = {};
  F8 acc[4];
  float mrow[8], lrow[8], alpha[8];
#pragma unroll
  for (int i = 0; i < 4; ++i) acc[i].v = zero;
#pragma unroll
  for (int v = 0; v < 8; ++v) { mrow[v] = -1e30f; lrow[v] = 0.f; }

  constexpr int NT = NSEQ / 64;  // 32 key tiles

#if ATTN_USE_TDM
  if (w == 0) {
    tdm_load_tile_64x64(kg, (unsigned)(size_t)(void*)&kbuf[0][0], DIN);
    tdm_load_tile_64x64(vg, (unsigned)(size_t)(void*)&vbuf[0][0], NSEQ);
  }
#endif

  for (int t = 0; t < NT; ++t) {
    const int kt = t * 64;
#if ATTN_USE_TDM
    const int bi = t & 1;
    if (w == 0) {
      if (t + 1 < NT) {
        tdm_load_tile_64x64(kg + (size_t)(kt + 64) * DIN,
                            (unsigned)(size_t)(void*)&kbuf[bi ^ 1][0], DIN);
        tdm_load_tile_64x64(vg + kt + 64,
                            (unsigned)(size_t)(void*)&vbuf[bi ^ 1][0], NSEQ);
        __builtin_amdgcn_s_wait_tensorcnt(2);  // tile t's two loads are done
      } else {
        __builtin_amdgcn_s_wait_tensorcnt(0);
      }
    }
    __syncthreads();
#else
    const int bi = 0;
    {  // cooperative synchronous staging: 4 threads per row, 16 elems each
      const int row = threadIdx.x >> 2;
      const int c   = (threadIdx.x & 3) * 16;
      *(v8bf*)(&kbuf[0][row * 64 + c]) =
          *(const v8bf*)(kg + (size_t)(kt + row) * DIN + c);
      *(v8bf*)(&kbuf[0][row * 64 + c + 8]) =
          *(const v8bf*)(kg + (size_t)(kt + row) * DIN + c + 8);
      *(v8bf*)(&vbuf[0][row * 64 + c]) =
          *(const v8bf*)(vg + (size_t)row * NSEQ + kt + c);
      *(v8bf*)(&vbuf[0][row * 64 + c + 8]) =
          *(const v8bf*)(vg + (size_t)row * NSEQ + kt + c + 8);
    }
    __syncthreads();
#endif
    const bf16* ktile = &kbuf[bi][0];
    const bf16* vtile = &vbuf[bi][0];

    // S = scale * Q K^T over 64 key columns (4 tiles of 16)
    F8 s[4];
#pragma unroll
    for (int ct = 0; ct < 4; ++ct) {
      const bf16* kcol = ktile + (ct * 16 + r) * 64 + hf * 8;
      v8f sv = zero;
      sv = wmma_bf16(aQ0, ld_fragB(kcol), sv);
      sv = wmma_bf16(aQ1, ld_fragB(kcol + 32), sv);
      s[ct].v = sv;
    }
#pragma unroll
    for (int ct = 0; ct < 4; ++ct)
#pragma unroll
      for (int v = 0; v < 8; ++v) s[ct].f[v] *= SCALE;

    // online softmax: per-row (row = v + hf*8) stats across 16 lanes
#pragma unroll
    for (int v = 0; v < 8; ++v) {
      float mx = fmaxf(fmaxf(s[0].f[v], s[1].f[v]),
                       fmaxf(s[2].f[v], s[3].f[v]));
#pragma unroll
      for (int off = 8; off >= 1; off >>= 1)
        mx = fmaxf(mx, __shfl_xor(mx, off, 16));
      const float mnew = fmaxf(mrow[v], mx);
      alpha[v] = __expf(mrow[v] - mnew);
      mrow[v]  = mnew;
      float rsum = 0.f;
#pragma unroll
      for (int ct = 0; ct < 4; ++ct) {
        const float p = __expf(s[ct].f[v] - mnew);
        s[ct].f[v] = p;
        rsum += p;
      }
#pragma unroll
      for (int off = 8; off >= 1; off >>= 1)
        rsum += __shfl_xor(rsum, off, 16);
      lrow[v] = lrow[v] * alpha[v] + rsum;
#pragma unroll
      for (int dt = 0; dt < 4; ++dt) acc[dt].f[v] *= alpha[v];
    }

    // P: C-layout -> LDS -> A-layout (wave-private tile, in-order DS ops)
#pragma unroll
    for (int ct = 0; ct < 4; ++ct)
#pragma unroll
      for (int v = 0; v < 8; ++v)
        pbuf[w][v + hf * 8][ct * 16 + r] = (bf16)s[ct].f[v];

    v16bf aP0 = ld_fragA(&pbuf[w][r][hf * 8]);
    v16bf aP1 = ld_fragA(&pbuf[w][r][32 + hf * 8]);

    // O += P (16x64) * V (64x64): vtile rows are K(=n)-contiguous columns
#pragma unroll
    for (int dt = 0; dt < 4; ++dt) {
      const bf16* vcol = vtile + (dt * 16 + r) * 64 + hf * 8;
      acc[dt].v = wmma_bf16(aP0, ld_fragB(vcol), acc[dt].v);
      acc[dt].v = wmma_bf16(aP1, ld_fragB(vcol + 32), acc[dt].v);
    }
    __syncthreads();  // all waves done with this buffer before it is refilled
  }

  // epilogue: normalize and store in scrambled reshape layout
#pragma unroll
  for (int v = 0; v < 8; ++v) {
    const float inv = 1.f / lrow[v];
    const int m = m0 + v + hf * 8;               // 0..2047
    const int rowp    = h * 128 + (m >> 4);
    const int colbase = (m & 15) * 64;
#pragma unroll
    for (int dt = 0; dt < 4; ++dt) {
      const int d = dt * 16 + r;
      Os[((size_t)(b * NSEQ + rowp)) * DIN + colbase + d] =
          (bf16)(acc[dt].f[v] * inv);
    }
  }
}

// ---------------- Host launcher ----------------

extern "C" void kernel_launch(void* const* d_in, const int* in_sizes, int n_in,
                              void* d_out, int out_size, void* d_ws,
                              size_t ws_size, hipStream_t stream) {
  (void)in_sizes; (void)n_in; (void)out_size; (void)ws_size;
  const float* q    = (const float*)d_in[0];
  const float* k    = (const float*)d_in[1];
  const float* v    = (const float*)d_in[2];
  const float* ln_g = (const float*)d_in[3];
  const float* ln_b = (const float*)d_in[4];
  const float* wq   = (const float*)d_in[5];
  const float* bq   = (const float*)d_in[6];
  const float* wk   = (const float*)d_in[7];
  const float* bk   = (const float*)d_in[8];
  const float* wv   = (const float*)d_in[9];
  const float* bv   = (const float*)d_in[10];
  const float* wo   = (const float*)d_in[11];
  const float* bo   = (const float*)d_in[12];

  char* ws = (char*)d_ws;
  const size_t ROWS   = (size_t)BATCH * NSEQ;             // 4096
  const size_t ACT_B  = ROWS * DIN * sizeof(bf16);        // 8 MB
  const size_t W_B    = (size_t)DIN * DIN * sizeof(bf16); // 2 MB

  bf16* Xn = (bf16*)(ws);                 // 3 * 8 MB (LN'd q,k,v)
  bf16* Wt = (bf16*)(ws + 3 * ACT_B);     // 4 * 2 MB (transposed weights)
  bf16* Qp = (bf16*)(ws + 3 * ACT_B + 4 * W_B);
  bf16* Kp = (bf16*)((char*)Qp + ACT_B);
  bf16* Vt = (bf16*)((char*)Kp + ACT_B);
  bf16* Os = Xn;  // reuse: LN buffers are dead once projections are done

  ln_cast_kernel<<<dim3(ROWS, 3), 256, 0, stream>>>(q, k, v, ln_g, ln_b, Xn);
  wcast_kernel<<<dim3((DIN * DIN) / 256, 4), 256, 0, stream>>>(wq, wk, wv, wo,
                                                               Wt);

  const dim3 ggrid(DIN / 128, ROWS / 128);  // (8, 32)
  gemm_kernel<<<ggrid, 128, 0, stream>>>(Xn, Wt, bq, Qp, 0);
  gemm_kernel<<<ggrid, 128, 0, stream>>>(Xn + ROWS * DIN,
                                         Wt + (size_t)DIN * DIN, bk, Kp, 0);
  gemm_kernel<<<ggrid, 128, 0, stream>>>(Xn + 2 * ROWS * DIN,
                                         Wt + 2 * (size_t)DIN * DIN, bv, Vt, 1);

  attn_kernel<<<dim3(NSEQ / 128, NHEAD, BATCH), 256, 0, stream>>>(Qp, Kp, Vt,
                                                                  Os);

  gemm_kernel<<<ggrid, 128, 0, stream>>>(Os, Wt + 3 * (size_t)DIN * DIN, bo,
                                         d_out, 2);
}
